// get_model_42167988912859
// MI455X (gfx1250) — matrix operand order repaired and checked by reading, not compiled, not run on hardware
//
#include <hip/hip_runtime.h>
#include <hip/hip_bf16.h>
#include <math.h>

// ---------------------------------------------------------------------------
// VN-PointNet part segmentation forward pass for MI455X (gfx1250).
// GEMM core: v_wmma_f32_16x16x32_bf16 (bf16 in, f32 accumulate),
// 64(M) x 128(N) block tile, double-buffered LDS (one barrier per K step),
// 4 waves x 8 accumulators; LDS staged in the ISA 7.12.2 per-lane fragment
// layout so fragments are fetched as ds_load_b128 pairs.
// ---------------------------------------------------------------------------

#define BB    8
#define NPTS  2048
#define KNN   20
#define PARTS 50
#define VN_EPS 1e-6f
#define BN_EPSF 1e-5f

static constexpr int C_O5CAT     = 1364;
static constexpr int ROWS_O5T    = 4092;   // 1364*3
static constexpr int ROWS_EXPAND = 4108;   // 4092 + 16 (label)
static constexpr int CAT_ROWS    = 9025;   // 4108 + 825 + 4092
static constexpr int OFF_O1234T  = 4108;
static constexpr int OFF_O5T     = 4933;

typedef __bf16        v16bf __attribute__((ext_vector_type(16)));
typedef float         v8f   __attribute__((ext_vector_type(8)));
typedef unsigned int  u32x4 __attribute__((ext_vector_type(4)));

union Frag { v16bf v; u32x4 u[2]; };

__device__ __forceinline__ unsigned short f2bf(float f) {
  unsigned int u = __float_as_uint(f);
  u += 0x7FFFu + ((u >> 16) & 1u);           // round-to-nearest-even
  return (unsigned short)(u >> 16);
}

// ---------------------------------------------------------------------------
// WMMA GEMM:  Out[b] = W(MxK) * X[b](KxN)  (+ optional bias per output row)
// Block = 128 threads = 4 waves. Block tile = 64(M) x 128(N); wave w owns rows
// [m0+16w, m0+16w+16) x 128 cols -> 8 16x16 WMMA accumulators per wave.
// Per 32-wide K step: stage A 64x32 + B 32x128 as bf16 into the ping-pong LDS
// buffer for step k+1 while computing step k; 32 WMMAs per block per K step.
// ---------------------------------------------------------------------------
__global__ void __launch_bounds__(128)
wmma_gemm_kernel(const float* __restrict__ W, const float* __restrict__ X,
                 float* __restrict__ Out, const float* __restrict__ bias,
                 int M, int Kd, long Ncols) {
  __shared__ __align__(16) unsigned short As[2][2048];  // 64x32 bf16 per buffer
  __shared__ __align__(16) unsigned short Bs[2][4096];  // 32x128 bf16 per buffer
  const int  b  = blockIdx.z;
  const int  m0 = blockIdx.y * 64;
  const long n0 = (long)blockIdx.x * 128;
  const float* Xb = X + (size_t)b * (size_t)Kd * (size_t)Ncols;
  float*       Ob = Out + (size_t)b * (size_t)M * (size_t)Ncols;
  const int tid  = threadIdx.x;
  const int wave = tid >> 5;
  const int lane = tid & 31;

  v8f acc[8];
#pragma unroll
  for (int ns = 0; ns < 8; ns++)
#pragma unroll
    for (int i = 0; i < 8; i++) acc[ns][i] = 0.f;

  const bool fullM = (m0 + 64 <= M);
  const bool fullN = (n0 + 128 <= Ncols);

  // Stage one 32-deep K slab into LDS buffer BUF, pre-permuted into the WMMA
  // per-lane fragment layout:
  //  A (per m-subtile): lane l (l2=l>>4,row=l&15): slot i   -> K = l2*8+i
  //                                                slot 8+i -> K = 16+l2*8+i
  //  B (per n-subtile): lane l (l2=l>>4,col=l&15): same K order as A
#define STAGE_TILE(K0, BUF)                                                     \
  do {                                                                          \
    const int  k0s = (K0);                                                      \
    const bool fk  = (k0s + 32 <= Kd);                                          \
    const bool fa  = fullM && fk;                                               \
    _Pragma("unroll")                                                           \
    for (int it = 0; it < 16; ++it) {                                           \
      int q = tid + 128 * it;            /* 0..2047 */                          \
      int row = q >> 5, kk = q & 31;                                            \
      float v = 0.f;                                                            \
      int gm = m0 + row, gk = k0s + kk;                                         \
      if (fa || (gm < M && gk < Kd)) v = W[(size_t)gm * Kd + gk];               \
      int msub = row >> 4, r = row & 15;                                        \
      int hi = kk >> 4, rem = kk & 15;                                          \
      As[BUF][msub * 512 + (r + 16 * (rem >> 3)) * 16 + (rem & 7) + 8 * hi] =   \
          f2bf(v);                                                              \
    }                                                                           \
    const bool fb = fullN && fk;                                                \
    _Pragma("unroll")                                                           \
    for (int it = 0; it < 32; ++it) {                                           \
      int q = tid + 128 * it;            /* 0..4095 */                          \
      int kk = q >> 7, nn = q & 127;                                            \
      float v = 0.f;                                                            \
      long gn = n0 + nn; int gk = k0s + kk;                                     \
      if (fb || (gk < Kd && gn < Ncols)) v = Xb[(size_t)gk * Ncols + gn];       \
      int ns = nn >> 4, col = nn & 15, kh = kk >> 4, e = kk & 15;               \
      Bs[BUF][ns * 512 + (col + 16 * (e >> 3)) * 16 + (e & 7) + 8 * kh] = f2bf(v); \
    }                                                                           \
  } while (0)

  const int nIter = (Kd + 31) / 32;
  STAGE_TILE(0, 0);
  for (int i = 0; i < nIter; ++i) {
    __syncthreads();   // buf[i&1] staged & prior reads of buf[(i+1)&1] done
    if (i + 1 < nIter) STAGE_TILE((i + 1) * 32, (i + 1) & 1);
    if (i + 2 < nIter) {
      long pn = n0 + (tid & 127);
      if (pn < Ncols)
        __builtin_prefetch(&Xb[(size_t)((i + 2) * 32) * Ncols + pn], 0, 1);
    }
    const int buf = i & 1;
    Frag af;
    const u32x4* ap =
        reinterpret_cast<const u32x4*>(&As[buf][wave * 512 + lane * 16]);
    af.u[0] = ap[0]; af.u[1] = ap[1];
#pragma unroll
    for (int ns = 0; ns < 8; ++ns) {
      Frag bf;
      const u32x4* bp =
          reinterpret_cast<const u32x4*>(&Bs[buf][ns * 512 + lane * 16]);
      bf.u[0] = bp[0]; bf.u[1] = bp[1];
      acc[ns] = __builtin_amdgcn_wmma_f32_16x16x32_bf16(false, af.v, false, bf.v,
                                                        (short)0, acc[ns], false, false);
    }
  }
#undef STAGE_TILE

  // C/D layout: VGPR r -> M = r + 8*(lane>=16), N = lane&15 (per 16-wide n tile)
  const int colL = lane & 15;
  const int mh   = (lane >> 4) * 8;
#pragma unroll
  for (int ns = 0; ns < 8; ++ns) {
    long gn = n0 + ns * 16 + colL;
    if (gn < Ncols) {
#pragma unroll
      for (int r = 0; r < 8; r++) {
        int gm = m0 + wave * 16 + mh + r;
        if (gm < M) {
          float v = acc[ns][r];
          if (bias) v += bias[gm];
          Ob[(size_t)gm * Ncols + gn] = v;
        }
      }
    }
  }
}

// ---------------------------------------------------------------------------
// Elementwise / reduction kernels
// ---------------------------------------------------------------------------
__global__ void zero_kernel(float* p, long n) {
  long i = (long)blockIdx.x * blockDim.x + threadIdx.x;
  if (i < n) p[i] = 0.f;
}

// kNN (largest 2*<xi,xj>-|xi|^2-|xj|^2 == nearest), one thread per point.
__global__ void knn_kernel(const float* __restrict__ pc, int* __restrict__ idx) {
  int t = blockIdx.x * blockDim.x + threadIdx.x;
  if (t >= BB * NPTS) return;
  int b = t / NPTS, n = t - b * NPTS;
  const float* p = pc + (size_t)b * 3 * NPTS;
  float x0 = p[n], y0 = p[NPTS + n], z0 = p[2 * NPTS + n];
  float best[KNN]; int bi[KNN];
#pragma unroll
  for (int j = 0; j < KNN; j++) { best[j] = -3.4e38f; bi[j] = 0; }
  for (int m = 0; m < NPTS; m++) {
    float dx = p[m] - x0, dy = p[NPTS + m] - y0, dz = p[2 * NPTS + m] - z0;
    float d = -(dx * dx + dy * dy + dz * dz);
    if (d > best[KNN - 1]) {
      int j = KNN - 1;
      while (j > 0 && best[j - 1] < d) {
        best[j] = best[j - 1]; bi[j] = bi[j - 1]; j--;
      }
      best[j] = d; bi[j] = m;
    }
  }
#pragma unroll
  for (int j = 0; j < KNN; j++) idx[(size_t)t * KNN + j] = bi[j];
}

// feat[b, s, c, n, j]: s=0 nbr-ctr, s=1 ctr, s=2 cross(nbr,ctr)
__global__ void graph_feat_kernel(const float* __restrict__ pc,
                                  const int* __restrict__ idx,
                                  float* __restrict__ feat) {
  long t = (long)blockIdx.x * blockDim.x + threadIdx.x;
  long tot = (long)BB * NPTS * KNN;
  if (t >= tot) return;
  int j = (int)(t % KNN);
  long bn = t / KNN;
  int n = (int)(bn % NPTS);
  int b = (int)(bn / NPTS);
  const float* p = pc + (size_t)b * 3 * NPTS;
  float cx = p[n], cy = p[NPTS + n], cz = p[2 * NPTS + n];
  int m = idx[t];
  float nx = p[m], ny = p[NPTS + m], nz = p[2 * NPTS + m];
  float rx = ny * cz - nz * cy;
  float ry = nz * cx - nx * cz;
  float rz = nx * cy - ny * cx;
  size_t SK = (size_t)NPTS * KNN;
  float* f = feat + (size_t)b * 9 * SK + (size_t)n * KNN + j;
  f[0 * SK] = nx - cx; f[1 * SK] = ny - cy; f[2 * SK] = nz - cz;
  f[3 * SK] = cx;      f[4 * SK] = cy;      f[5 * SK] = cz;
  f[6 * SK] = rx;      f[7 * SK] = ry;      f[8 * SK] = rz;
}

// VN batchnorm: stats of n=|v|+eps per channel over (batch, spatial).
__global__ void vnbn_stats_kernel(const float* __restrict__ x,
                                  float* __restrict__ stats, int C, long S) {
  int c = blockIdx.x;
  __shared__ float s1[256], s2[256];
  long tot = (long)BB * S;
  float a1 = 0.f, a2 = 0.f;
  for (long i = (long)blockIdx.y * blockDim.x + threadIdx.x; i < tot;
       i += (long)gridDim.y * blockDim.x) {
    long b = i / S, s = i - b * S;
    const float* p = x + (((size_t)b * C + c) * 3) * S + s;
    float n = sqrtf(p[0] * p[0] + p[S] * p[S] + p[2 * S] * p[2 * S]) + VN_EPS;
    a1 += n; a2 += n * n;
  }
  s1[threadIdx.x] = a1; s2[threadIdx.x] = a2;
  __syncthreads();
  for (int o = 128; o > 0; o >>= 1) {
    if ((int)threadIdx.x < o) {
      s1[threadIdx.x] += s1[threadIdx.x + o];
      s2[threadIdx.x] += s2[threadIdx.x + o];
    }
    __syncthreads();
  }
  if (threadIdx.x == 0) {
    atomicAdd(&stats[c], s1[0]);
    atomicAdd(&stats[C + c], s2[0]);
  }
}

__global__ void vnbn_apply_kernel(const float* __restrict__ x,
                                  float* __restrict__ out,
                                  const float* __restrict__ stats,
                                  const float* __restrict__ g,
                                  const float* __restrict__ be,
                                  int C, long S) {
  long t = (long)blockIdx.x * blockDim.x + threadIdx.x;
  long tot = (long)BB * C * S;
  if (t >= tot) return;
  long s = t % S; long bc = t / S;
  int c = (int)(bc % C); int b = (int)(bc / C);
  const float* p = x + (((size_t)b * C + c) * 3) * S + s;
  float x0 = p[0], x1 = p[S], x2 = p[2 * S];
  float n = sqrtf(x0 * x0 + x1 * x1 + x2 * x2) + VN_EPS;
  float cnt = (float)((long)BB * S);
  float m = stats[c] / cnt;
  float v = stats[C + c] / cnt - m * m;
  float nbn = (n - m) * rsqrtf(v + BN_EPSF) * g[c] + be[c];
  float sc = nbn / n;
  float* q = out + (((size_t)b * C + c) * 3) * S + s;
  q[0] = x0 * sc; q[S] = x1 * sc; q[2 * S] = x2 * sc;
}

// VN leaky-relu (slope 0): out = dot(q,d)>=0 ? q : q - (dot/(|d|^2+eps)) d
__global__ void vn_combine_kernel(const float* __restrict__ q,
                                  const float* __restrict__ d,
                                  float* __restrict__ out, int C, long S) {
  long t = (long)blockIdx.x * blockDim.x + threadIdx.x;
  long tot = (long)BB * C * S;
  if (t >= tot) return;
  long s = t % S; long bc = t / S;
  int c = (int)(bc % C); int b = (int)(bc / C);
  size_t off = (((size_t)b * C + c) * 3) * S + s;
  float q0 = q[off], q1 = q[off + S], q2 = q[off + 2 * S];
  float d0 = d[off], d1 = d[off + S], d2 = d[off + 2 * S];
  float dot = q0 * d0 + q1 * d1 + q2 * d2;
  float dsq = d0 * d0 + d1 * d1 + d2 * d2;
  float f = (dot >= 0.f) ? 0.f : dot / (dsq + VN_EPS);
  out[off]         = q0 - f * d0;
  out[off + S]     = q1 - f * d1;
  out[off + 2 * S] = q2 - f * d2;
}

// VN maxpool: x,d: [B,C,3,So,L] -> out [B,C,3,So], argmax_l dot3(x,d)
__global__ void vn_maxpool_kernel(const float* __restrict__ x,
                                  const float* __restrict__ d,
                                  float* __restrict__ out,
                                  int C, long So, int L) {
  long t = (long)blockIdx.x * blockDim.x + threadIdx.x;
  long tot = (long)BB * C * So;
  if (t >= tot) return;
  long so = t % So; long bc = t / So;
  int c = (int)(bc % C); int b = (int)(bc / C);
  size_t vs = (size_t)So * L;
  size_t base = (((size_t)b * C + c) * 3 * So + so) * (size_t)L;
  float best = -3.4e38f; int biL = 0;
  for (int l = 0; l < L; l++) {
    float a = x[base + l] * d[base + l]
            + x[base + vs + l] * d[base + vs + l]
            + x[base + 2 * vs + l] * d[base + 2 * vs + l];
    if (a > best) { best = a; biL = l; }
  }
  size_t ob = ((size_t)b * C + c) * 3 * So + so;
#pragma unroll
  for (int v = 0; v < 3; v++)
    out[ob + (size_t)v * So] = x[base + (size_t)v * vs + biL];
}

// src [B,Csrc,S] -> dst [B,Cdst,S] rows [rowOff, rowOff+Csrc)
__global__ void copy_rows_kernel(const float* __restrict__ src,
                                 float* __restrict__ dst,
                                 int Csrc, int Cdst, int rowOff, long S) {
  long t = (long)blockIdx.x * blockDim.x + threadIdx.x;
  long tot = (long)BB * Csrc * S;
  if (t >= tot) return;
  long s = t % S; long bc = t / S;
  int c = (int)(bc % Csrc); int b = (int)(bc / Csrc);
  dst[((size_t)b * Cdst + rowOff + c) * S + s] =
      src[((size_t)b * Csrc + c) * S + s];
}

// src [B,Csrc,3] -> dst [B,Cdst,3,N] broadcast over N
__global__ void bcast_vec3_kernel(const float* __restrict__ src,
                                  float* __restrict__ dst,
                                  int Csrc, int Cdst, int rowOff, long Nn) {
  long t = (long)blockIdx.x * blockDim.x + threadIdx.x;
  long tot = (long)BB * Csrc * 3 * Nn;
  if (t >= tot) return;
  long n = t % Nn; long r = t / Nn;
  int v = (int)(r % 3); long bc = r / 3;
  int c = (int)(bc % Csrc); int b = (int)(bc / Csrc);
  dst[(((size_t)b * Cdst + rowOff + c) * 3 + v) * Nn + n] =
      src[((size_t)b * Csrc + c) * 3 + v];
}

// x [B,C,3,N] -> out [B,C,3] mean over N
__global__ void mean_last_kernel(const float* __restrict__ x,
                                 float* __restrict__ out, int C, long Nn) {
  long t = (long)blockIdx.x * blockDim.x + threadIdx.x;
  long tot = (long)BB * C * 3;
  if (t >= tot) return;
  const float* p = x + (size_t)t * Nn;
  float s = 0.f;
  for (long n = 0; n < Nn; n++) s += p[n];
  out[t] = s / (float)Nn;
}

// out[b, rowOff+i*3+k, n] = sum_j x[b,i,j,n] * zl[b,k,j,n]   (trans = swapaxes(zl))
__global__ void vn_transform_kernel(const float* __restrict__ x,
                                    const float* __restrict__ zl,
                                    float* __restrict__ out,
                                    int C, int outRows, int rowOff, long Nn) {
  long t = (long)blockIdx.x * blockDim.x + threadIdx.x;
  long tot = (long)BB * C * Nn;
  if (t >= tot) return;
  long n = t % Nn; long bi = t / Nn;
  int i = (int)(bi % C); int b = (int)(bi / C);
  const float* xp = x + (((size_t)b * C + i) * 3) * Nn + n;
  const float* zp = zl + (size_t)b * 9 * Nn + n;
  float x0 = xp[0], x1 = xp[Nn], x2 = xp[2 * Nn];
#pragma unroll
  for (int k = 0; k < 3; k++) {
    float v = x0 * zp[(size_t)(k * 3 + 0) * Nn]
            + x1 * zp[(size_t)(k * 3 + 1) * Nn]
            + x2 * zp[(size_t)(k * 3 + 2) * Nn];
    out[((size_t)b * outRows + rowOff + i * 3 + k) * Nn + n] = v;
  }
}

__global__ void rowmax_kernel(const float* __restrict__ cat,
                              float* __restrict__ omax,
                              int rowOff, int rows, long Nn) {
  long t = (long)blockIdx.x * blockDim.x + threadIdx.x;
  long tot = (long)BB * rows;
  if (t >= tot) return;
  int r = (int)(t % rows); int b = (int)(t / rows);
  const float* p = cat + ((size_t)b * CAT_ROWS + rowOff + r) * Nn;
  float m = -3.4e38f;
  for (long n = 0; n < Nn; n++) m = fmaxf(m, p[n]);
  omax[(size_t)b * ROWS_EXPAND + r] = m;
}

__global__ void label_omax_kernel(const float* __restrict__ label,
                                  float* __restrict__ omax) {
  int t = blockIdx.x * blockDim.x + threadIdx.x;
  if (t >= BB * 16) return;
  int b = t / 16, i = t % 16;
  omax[(size_t)b * ROWS_EXPAND + ROWS_O5T + i] = label[b * 16 + i];
}

__global__ void expand_kernel(const float* __restrict__ omax,
                              float* __restrict__ cat, long Nn) {
  long t = (long)blockIdx.x * blockDim.x + threadIdx.x;
  long tot = (long)BB * ROWS_EXPAND * Nn;
  if (t >= tot) return;
  long n = t % Nn; long br = t / Nn;
  int r = (int)(br % ROWS_EXPAND); int b = (int)(br / ROWS_EXPAND);
  cat[((size_t)b * CAT_ROWS + r) * Nn + n] = omax[(size_t)b * ROWS_EXPAND + r];
}

// BatchNorm1d over (B,N) per channel
__global__ void bn1d_stats_kernel(const float* __restrict__ x,
                                  float* __restrict__ stats, int C, long Nn) {
  int c = blockIdx.x;
  __shared__ float s1[256], s2[256];
  long tot = (long)BB * Nn;
  float a1 = 0.f, a2 = 0.f;
  for (long i = (long)blockIdx.y * blockDim.x + threadIdx.x; i < tot;
       i += (long)gridDim.y * blockDim.x) {
    long b = i / Nn, n = i - b * Nn;
    float v = x[((size_t)b * C + c) * Nn + n];
    a1 += v; a2 += v * v;
  }
  s1[threadIdx.x] = a1; s2[threadIdx.x] = a2;
  __syncthreads();
  for (int o = 128; o > 0; o >>= 1) {
    if ((int)threadIdx.x < o) {
      s1[threadIdx.x] += s1[threadIdx.x + o];
      s2[threadIdx.x] += s2[threadIdx.x + o];
    }
    __syncthreads();
  }
  if (threadIdx.x == 0) {
    atomicAdd(&stats[c], s1[0]);
    atomicAdd(&stats[C + c], s2[0]);
  }
}

__global__ void bn1d_apply_kernel(float* __restrict__ x,
                                  const float* __restrict__ stats,
                                  const float* __restrict__ g,
                                  const float* __restrict__ be,
                                  int C, long Nn, int relu) {
  long t = (long)blockIdx.x * blockDim.x + threadIdx.x;
  long tot = (long)BB * C * Nn;
  if (t >= tot) return;
  long n = t % Nn; long bc = t / Nn;
  int c = (int)(bc % C); int b = (int)(bc / C);
  float cnt = (float)((long)BB * Nn);
  float m = stats[c] / cnt;
  float v = stats[C + c] / cnt - m * m;
  size_t off = ((size_t)b * C + c) * Nn + n;
  float y = (x[off] - m) * rsqrtf(v + BN_EPSF) * g[c] + be[c];
  if (relu) y = fmaxf(y, 0.f);
  x[off] = y;
}

// h [B,50,N] -> out [B,N,50] log_softmax over channels
__global__ void logsoftmax_kernel(const float* __restrict__ h,
                                  float* __restrict__ out) {
  int t = blockIdx.x * blockDim.x + threadIdx.x;
  if (t >= BB * NPTS) return;
  int n = t % NPTS, b = t / NPTS;
  const float* p = h + (size_t)b * PARTS * NPTS + n;
  float mx = -3.4e38f;
  for (int c = 0; c < PARTS; c++) mx = fmaxf(mx, p[(size_t)c * NPTS]);
  float s = 0.f;
  for (int c = 0; c < PARTS; c++) s += expf(p[(size_t)c * NPTS] - mx);
  float ls = logf(s);
  float* o = out + ((size_t)b * NPTS + n) * PARTS;
  for (int c = 0; c < PARTS; c++) o[c] = p[(size_t)c * NPTS] - mx - ls;
}

// ---------------------------------------------------------------------------
// Host-side orchestration
// ---------------------------------------------------------------------------
namespace {

struct Arena {
  char* base; size_t off; size_t cap;
  float* f(size_t n) {
    size_t bytes = ((n * sizeof(float)) + 255) & ~(size_t)255;
    float* p = (float*)(base + off);
    off += bytes;
    return p;
  }
  int* ip(size_t n) { return (int*)f(n); }
};

inline int nblk(long n, int bs = 256) { return (int)((n + bs - 1) / bs); }

void run_gemm(hipStream_t st, const float* W, const float* X, float* Out,
              const float* bias, int M, int Kd, long Ncols) {
  dim3 g((unsigned)((Ncols + 127) / 128), (unsigned)((M + 63) / 64), BB);
  wmma_gemm_kernel<<<g, dim3(128), 0, st>>>(W, X, Out, bias, M, Kd, Ncols);
}

void run_vnbn(hipStream_t st, const float* x, float* out, float* stats,
              const float* g, const float* be, int C, long S) {
  zero_kernel<<<nblk(2 * C), 256, 0, st>>>(stats, 2 * C);
  long tot = (long)BB * S;
  int ch = (int)((tot + 255) / 256); if (ch > 128) ch = 128; if (ch < 1) ch = 1;
  vnbn_stats_kernel<<<dim3((unsigned)C, (unsigned)ch), 256, 0, st>>>(x, stats, C, S);
  long te = (long)BB * C * S;
  vnbn_apply_kernel<<<nblk(te), 256, 0, st>>>(x, out, stats, g, be, C, S);
}

// VNLinearLeakyReLU: q = bn(Wf x); d = Wd x; out = leaky-project(q, d)
void run_vn_lrelu(hipStream_t st, const float* Wf, const float* Wd,
                  const float* g, const float* be, const float* x, float* out,
                  float* tq, float* td, float* stats, int Cin, int Cout, long S) {
  run_gemm(st, Wf, x, tq, nullptr, Cout, Cin, 3 * S);
  run_vnbn(st, tq, tq, stats, g, be, Cout, S);
  run_gemm(st, Wd, x, td, nullptr, Cout, Cin, 3 * S);
  long te = (long)BB * Cout * S;
  vn_combine_kernel<<<nblk(te), 256, 0, st>>>(tq, td, out, Cout, S);
}

void run_vn_maxpool(hipStream_t st, const float* Wd, const float* x,
                    float* dtmp, float* out, int C, long So, int L) {
  run_gemm(st, Wd, x, dtmp, nullptr, C, C, 3 * So * L);
  long tot = (long)BB * C * So;
  vn_maxpool_kernel<<<nblk(tot), 256, 0, st>>>(x, dtmp, out, C, So, L);
}

void run_bn1d(hipStream_t st, float* x, float* stats, const float* g,
              const float* be, int C, long Nn, int relu) {
  zero_kernel<<<nblk(2 * C), 256, 0, st>>>(stats, 2 * C);
  long tot = (long)BB * Nn;
  int ch = (int)((tot + 255) / 256); if (ch > 128) ch = 128; if (ch < 1) ch = 1;
  bn1d_stats_kernel<<<dim3((unsigned)C, (unsigned)ch), 256, 0, st>>>(x, stats, C, Nn);
  long te = (long)BB * C * Nn;
  bn1d_apply_kernel<<<nblk(te), 256, 0, st>>>(x, stats, g, be, C, Nn, relu);
}

} // namespace

// Input leaf order assumed = setup_inputs() dict insertion order, depth-first.
enum {
  IN_PC = 0, IN_LABEL,
  CP_WF, CP_WD, CP_G, CP_B,
  POOL_WD,
  C1_WF, C1_WD, C1_G, C1_B,
  C2_WF, C2_WD, C2_G, C2_B,
  C3_WF, C3_WD, C3_G, C3_B,
  C4_WF, C4_WD, C4_G, C4_B,
  C5_W, BN5_G, BN5_B,
  F_C1_WF, F_C1_WD, F_C1_G, F_C1_B,
  F_C2_WF, F_C2_WD, F_C2_G, F_C2_B,
  F_C3_WF, F_C3_WD, F_C3_G, F_C3_B,
  F_POOL,
  F_FC1_WF, F_FC1_WD, F_FC1_G, F_FC1_B,
  F_FC2_WF, F_FC2_WD, F_FC2_G, F_FC2_B,
  F_FC3,
  S_V1_WF, S_V1_WD, S_V1_G, S_V1_B,
  S_V2_WF, S_V2_WD, S_V2_G, S_V2_B,
  S_WLIN,
  H_W1, H_B1, H_G1, H_BE1,
  H_W2, H_B2, H_G2, H_BE2,
  H_W3, H_B3, H_G3, H_BE3,
  H_W4, H_B4
};

extern "C" void kernel_launch(void* const* d_in, const int* in_sizes, int n_in,
                              void* d_out, int out_size, void* d_ws, size_t ws_size,
                              hipStream_t stream) {
  (void)in_sizes; (void)n_in; (void)out_size;
  auto F = [&](int i) { return (const float*)d_in[i]; };
  const float* pc    = F(IN_PC);
  const float* label = F(IN_LABEL);

  Arena A{(char*)d_ws, 0, ws_size};
  const long Nn = NPTS, N3 = 3L * NPTS, SK = (long)NPTS * KNN;

  float* stats  = A.f(2048);
  int*   knnidx = A.ip((size_t)BB * NPTS * KNN);
  float* feat   = A.f((size_t)BB * 9 * SK);
  float* tq     = A.f((size_t)BB * 682 * N3);     // largest GEMM temp
  float* td     = A.f((size_t)BB * 682 * N3);
  float* ta     = A.f((size_t)BB * 682 * N3);     // stage outputs (conv_pos, z1)
  float* tb     = A.f((size_t)BB * 341 * N3);     // z2 / fstn s3
  float* tc1    = A.f((size_t)BB * 21 * N3);      // fstn s1
  float* tc2    = A.f((size_t)BB * 42 * N3);      // fstn s2
  float* xp     = A.f((size_t)BB * 21 * N3);
  float* o1     = A.f((size_t)BB * 21 * N3);
  float* o2     = A.f((size_t)BB * 42 * N3);
  float* o3     = A.f((size_t)BB * 42 * N3);
  float* o4     = A.f((size_t)BB * 170 * N3);
  float* o5     = A.f((size_t)BB * 682 * N3);
  float* o5c    = A.f((size_t)BB * C_O5CAT * N3);
  float* mean5  = A.f((size_t)BB * 682 * 3);
  float* pooled = A.f((size_t)BB * 341 * 3);
  float* f1b    = A.f((size_t)BB * 170 * 3);
  float* f2b    = A.f((size_t)BB * 85 * 3);
  float* gb     = A.f((size_t)BB * 42 * 3);
  float* zlin   = A.f((size_t)BB * 9 * NPTS);
  float* o1234  = A.f((size_t)BB * 275 * N3);
  float* omaxb  = A.f((size_t)BB * ROWS_EXPAND);
  float* catb   = A.f((size_t)BB * CAT_ROWS * NPTS);
  float* h1     = A.f((size_t)BB * 256 * NPTS);
  float* h2     = A.f((size_t)BB * 256 * NPTS);
  float* h3     = A.f((size_t)BB * 128 * NPTS);
  float* h4     = A.f((size_t)BB * PARTS * NPTS);
  float* nt     = A.f((size_t)BB * 84 * N3);

  // ---- graph features ----
  knn_kernel<<<nblk((long)BB * NPTS), 256, 0, stream>>>(pc, knnidx);
  graph_feat_kernel<<<nblk((long)BB * NPTS * KNN), 256, 0, stream>>>(pc, knnidx, feat);

  // ---- conv_pos (3 -> 21 over N*K) + pool over K ----
  run_vn_lrelu(stream, F(CP_WF), F(CP_WD), F(CP_G), F(CP_B), feat, ta, tq, td,
               stats, 3, 21, SK);
  run_vn_maxpool(stream, F(POOL_WD), ta, td, xp, 21, NPTS, KNN);

  // ---- conv1..conv3 ----
  run_vn_lrelu(stream, F(C1_WF), F(C1_WD), F(C1_G), F(C1_B), xp, o1, tq, td, stats, 21, 21, Nn);
  run_vn_lrelu(stream, F(C2_WF), F(C2_WD), F(C2_G), F(C2_B), o1, o2, tq, td, stats, 21, 42, Nn);
  run_vn_lrelu(stream, F(C3_WF), F(C3_WD), F(C3_G), F(C3_B), o2, o3, tq, td, stats, 42, 42, Nn);

  // ---- feature STN on o3 ----
  run_vn_lrelu(stream, F(F_C1_WF), F(F_C1_WD), F(F_C1_G), F(F_C1_B), o3, tc1, tq, td, stats, 42, 21, Nn);
  run_vn_lrelu(stream, F(F_C2_WF), F(F_C2_WD), F(F_C2_G), F(F_C2_B), tc1, tc2, tq, td, stats, 21, 42, Nn);
  run_vn_lrelu(stream, F(F_C3_WF), F(F_C3_WD), F(F_C3_G), F(F_C3_B), tc2, tb, tq, td, stats, 42, 341, Nn);
  run_vn_maxpool(stream, F(F_POOL), tb, td, pooled, 341, 1, NPTS);
  run_vn_lrelu(stream, F(F_FC1_WF), F(F_FC1_WD), F(F_FC1_G), F(F_FC1_B), pooled, f1b, tq, td, stats, 341, 170, 1);
  run_vn_lrelu(stream, F(F_FC2_WF), F(F_FC2_WD), F(F_FC2_G), F(F_FC2_B), f1b, f2b, tq, td, stats, 170, 85, 1);
  run_gemm(stream, F(F_FC3), f2b, gb, nullptr, 42, 85, 3);

  // ---- nt = concat(o3, broadcast g); conv4; conv5 + bn ----
  copy_rows_kernel<<<nblk((long)BB * 42 * N3), 256, 0, stream>>>(o3, nt, 42, 84, 0, N3);
  bcast_vec3_kernel<<<nblk((long)BB * 42 * 3 * Nn), 256, 0, stream>>>(gb, nt, 42, 84, 42, Nn);
  run_vn_lrelu(stream, F(C4_WF), F(C4_WD), F(C4_G), F(C4_B), nt, o4, tq, td, stats, 84, 170, Nn);
  run_gemm(stream, F(C5_W), o4, o5, nullptr, 682, 170, N3);
  run_vnbn(stream, o5, o5, stats, F(BN5_G), F(BN5_B), 682, Nn);

  // ---- o5cat = concat(o5, mean_N(o5)) ----
  mean_last_kernel<<<nblk((long)BB * 682 * 3), 256, 0, stream>>>(o5, mean5, 682, Nn);
  copy_rows_kernel<<<nblk((long)BB * 682 * N3), 256, 0, stream>>>(o5, o5c, 682, C_O5CAT, 0, N3);
  bcast_vec3_kernel<<<nblk((long)BB * 682 * 3 * Nn), 256, 0, stream>>>(mean5, o5c, 682, C_O5CAT, 682, Nn);

  // ---- std feature: z1, z2, zlin; transform o5cat into cat rows ----
  run_vn_lrelu(stream, F(S_V1_WF), F(S_V1_WD), F(S_V1_G), F(S_V1_B), o5c, ta, tq, td, stats, C_O5CAT, 682, Nn);
  run_vn_lrelu(stream, F(S_V2_WF), F(S_V2_WD), F(S_V2_G), F(S_V2_B), ta, tb, tq, td, stats, 682, 341, Nn);
  run_gemm(stream, F(S_WLIN), tb, zlin, nullptr, 3, 341, N3);
  vn_transform_kernel<<<nblk((long)BB * C_O5CAT * Nn), 256, 0, stream>>>(
      o5c, zlin, catb, C_O5CAT, CAT_ROWS, OFF_O5T, Nn);

  // ---- global max + label -> expanded rows ----
  rowmax_kernel<<<nblk((long)BB * ROWS_O5T), 256, 0, stream>>>(catb, omaxb, OFF_O5T, ROWS_O5T, Nn);
  label_omax_kernel<<<nblk(BB * 16), 256, 0, stream>>>(label, omaxb);
  expand_kernel<<<nblk((long)BB * ROWS_EXPAND * Nn), 256, 0, stream>>>(omaxb, catb, Nn);

  // ---- o1234 concat + transform into cat rows ----
  copy_rows_kernel<<<nblk((long)BB * 21 * N3), 256, 0, stream>>>(o1, o1234, 21, 275, 0, N3);
  copy_rows_kernel<<<nblk((long)BB * 42 * N3), 256, 0, stream>>>(o2, o1234, 42, 275, 21, N3);
  copy_rows_kernel<<<nblk((long)BB * 42 * N3), 256, 0, stream>>>(o3, o1234, 42, 275, 63, N3);
  copy_rows_kernel<<<nblk((long)BB * 170 * N3), 256, 0, stream>>>(o4, o1234, 170, 275, 105, N3);
  vn_transform_kernel<<<nblk((long)BB * 275 * Nn), 256, 0, stream>>>(
      o1234, zlin, catb, 275, CAT_ROWS, OFF_O1234T, Nn);

  // ---- head MLP ----
  run_gemm(stream, F(H_W1), catb, h1, F(H_B1), 256, CAT_ROWS, Nn);
  run_bn1d(stream, h1, stats, F(H_G1), F(H_BE1), 256, Nn, 1);
  run_gemm(stream, F(H_W2), h1, h2, F(H_B2), 256, 256, Nn);
  run_bn1d(stream, h2, stats, F(H_G2), F(H_BE2), 256, Nn, 1);
  run_gemm(stream, F(H_W3), h2, h3, F(H_B3), 128, 256, Nn);
  run_bn1d(stream, h3, stats, F(H_G3), F(H_BE3), 128, Nn, 1);
  run_gemm(stream, F(H_W4), h3, h4, F(H_B4), PARTS, 128, Nn);

  logsoftmax_kernel<<<nblk((long)BB * NPTS), 256, 0, stream>>>(h4, (float*)d_out);
}
